// BlendEmoBackbone_36593121362122
// MI455X (gfx1250) — matrix operand order repaired and checked by reading, not compiled, not run on hardware
//
#include <hip/hip_runtime.h>
#include <hip/hip_bf16.h>
#include <math.h>

// ---------------------------------------------------------------------------
// BlendEmoBackbone for MI455X (gfx1250, wave32, WMMA bf16 f32-accum)
// B=4096, M=4, H=1024, R=10, MID=512, D3=3072, K_aug padded 1025 -> 1056
// ---------------------------------------------------------------------------

#define B_ROWS 4096
#define H_DIM  1024
#define D3_DIM 3072
#define MID_DIM 512
#define R_RANK 10
#define KAUG   1056   // 1025 padded up to multiple of 32
#define BPITCH 40     // LDS row pitch (elements): 80B, 16B-aligned, 20-bank stride

typedef __attribute__((ext_vector_type(8)))  __bf16 v8bf;
typedef __attribute__((ext_vector_type(16))) __bf16 v16bf;
typedef __attribute__((ext_vector_type(8)))  float  v8f;

// ---------------------------- device helpers -------------------------------

__device__ __forceinline__ float gelu_exact(float x) {
    return 0.5f * x * (1.0f + erff(x * 0.70710678118654752f));
}

__device__ __forceinline__ float block_reduce_sum(float v, float* sbuf) {
    // wave32 reduce
    for (int o = 16; o > 0; o >>= 1) v += __shfl_xor(v, o, 32);
    int wid = threadIdx.x >> 5;
    if ((threadIdx.x & 31) == 0) sbuf[wid] = v;
    __syncthreads();
    int nw = blockDim.x >> 5;
    float s = (threadIdx.x < nw) ? sbuf[threadIdx.x] : 0.0f;
    for (int o = 16; o > 0; o >>= 1) s += __shfl_xor(s, o, 32);
    if (threadIdx.x == 0) sbuf[0] = s;
    __syncthreads();
    float r = sbuf[0];
    __syncthreads();
    return r;
}

// CDNA5 async copy: global -> LDS, 16B per lane, tracked by ASYNCcnt.
__device__ __forceinline__ void async_copy_b128(unsigned int lds_off,
                                                const __bf16* gaddr) {
    asm volatile("global_load_async_to_lds_b128 %0, %1, off"
                 :: "v"(lds_off), "v"(gaddr) : "memory");
}
__device__ __forceinline__ void wait_async0() {
    asm volatile("s_wait_asynccnt 0x0" ::: "memory");
}

// A fragment: 16x32 bf16 tile, A row-major MxK (lda in elements).
// lanes 0-15 hold M=0..15 with K 0..15;
// lanes 16-31 hold M=0..15 with K 16..31.
__device__ __forceinline__ v16bf load_afrag(const __bf16* __restrict__ A,
                                            int lda, int row0, int kb) {
    int l = threadIdx.x & 31;
    int m = l & 15;
    int base = (l < 16) ? 0 : 16;
    const __bf16* p = A + (size_t)(row0 + m) * (size_t)lda + kb + base;
    v8bf lo = *(const v8bf*)p;          // K = base .. base+7
    v8bf hi = *(const v8bf*)(p + 8);    // K = base+8 .. base+15
    v16bf r;
    #pragma unroll
    for (int i = 0; i < 8; i++) { r[i] = lo[i]; r[8 + i] = hi[i]; }
    return r;
}

// B fragment: 32x16 bf16 (KxN), source stored as weights N-major: Bw[n][k]
// (ldb in elements). lanes 0-15: K=0..15 of column n; lanes 16-31: K=16..31.
__device__ __forceinline__ v16bf load_bfrag(const __bf16* __restrict__ Bw,
                                            int ldb, int col0, int kb) {
    int l = threadIdx.x & 31;
    int n = l & 15;
    int koff = (l < 16) ? 0 : 16;
    const __bf16* p = Bw + (size_t)(col0 + n) * (size_t)ldb + kb + koff;
    v8bf lo = *(const v8bf*)p;
    v8bf hi = *(const v8bf*)(p + 8);
    v16bf r;
    #pragma unroll
    for (int i = 0; i < 8; i++) { r[i] = lo[i]; r[8 + i] = hi[i]; }
    return r;
}

// B fragment from LDS panel (n-major, pitch BPITCH elements).
__device__ __forceinline__ v16bf lds_bfrag(const __bf16* base, int nsub) {
    int l = threadIdx.x & 31;
    int n = (l & 15) + nsub;
    int koff = (l < 16) ? 0 : 16;
    const __bf16* p = base + n * BPITCH + koff;
    v8bf lo = *(const v8bf*)p;
    v8bf hi = *(const v8bf*)(p + 8);
    v16bf r;
    #pragma unroll
    for (int i = 0; i < 8; i++) { r[i] = lo[i]; r[8 + i] = hi[i]; }
    return r;
}

// ------------------------------ conversion ---------------------------------

__global__ void cvt_bf16_kernel(const float* __restrict__ src,
                                __bf16* __restrict__ dst, size_t n) {
    size_t i = (size_t)blockIdx.x * blockDim.x + threadIdx.x;
    size_t stride = (size_t)gridDim.x * blockDim.x;
    for (; i < n; i += stride) dst[i] = (__bf16)src[i];
}

// factors (4,10,1025,1024) f32  ->  fbT (4*10, 1024, KAUG) bf16, N-major, K padded
__global__ void cvt_factors_kernel(const float* __restrict__ f,
                                   __bf16* __restrict__ dst) {
    const size_t n = (size_t)40 * H_DIM * KAUG;
    size_t i = (size_t)blockIdx.x * blockDim.x + threadIdx.x;
    size_t stride = (size_t)gridDim.x * blockDim.x;
    for (; i < n; i += stride) {
        size_t mr  = i / ((size_t)H_DIM * KAUG);
        size_t rem = i % ((size_t)H_DIM * KAUG);
        int h = (int)(rem / KAUG);
        int k = (int)(rem % KAUG);
        float v = (k < 1025) ? f[(mr * 1025 + (size_t)k) * H_DIM + h] : 0.0f;
        dst[i] = (__bf16)v;
    }
}

__global__ void audio_bf16_kernel(const float* __restrict__ tokens,
                                  __bf16* __restrict__ dst) {
    size_t n = (size_t)B_ROWS * H_DIM;
    size_t i = (size_t)blockIdx.x * blockDim.x + threadIdx.x;
    size_t stride = (size_t)gridDim.x * blockDim.x;
    for (; i < n; i += stride) {
        size_t b = i >> 10;
        int h = (int)(i & 1023);
        dst[i] = (__bf16)tokens[(b * 4 + 1) * H_DIM + h];
    }
}

// ------------------------- generic bf16 WMMA GEMM --------------------------
// C[M,N] = act(A[M,K] @ Bw[N,K]^T + bias). grid=(M/128, N/16), block=256.
__global__ void gemm_bf16_nt(const __bf16* __restrict__ A, int lda,
                             const __bf16* __restrict__ Bw, int ldb,
                             float* __restrict__ C, int ldc,
                             const float* __restrict__ bias, int act, int K) {
    int wave = threadIdx.x >> 5;
    int lane = threadIdx.x & 31;
    int row0 = blockIdx.x * 128 + wave * 16;
    int col0 = blockIdx.y * 16;
    v8f acc = {};
    for (int kb = 0; kb < K; kb += 32) {
        v16bf a = load_afrag(A, lda, row0, kb);
        v16bf b = load_bfrag(Bw, ldb, col0, kb);
        acc = __builtin_amdgcn_wmma_f32_16x16x32_bf16(false, a, false, b,
                                                      (short)0, acc, false, false);
    }
    int col = col0 + (lane & 15);
    float bv = bias ? bias[col] : 0.0f;
    #pragma unroll
    for (int j = 0; j < 8; j++) {
        int row = row0 + ((lane < 16) ? j : 8 + j);
        float v = acc[j] + bv;
        if (act == 1) v = gelu_exact(v);
        C[(size_t)row * ldc + col] = v;
    }
}

// ------------------------- gate feature builders ---------------------------

// feat_o[(b*3+j)] = LN(concat(others_j, audio, |others_j - audio|)), bf16
__global__ void feat_others_kernel(const float* __restrict__ tokens,
                                   const int* __restrict__ tmask,
                                   const float* __restrict__ lnw,
                                   const float* __restrict__ lnb,
                                   __bf16* __restrict__ feat) {
    __shared__ float sbuf[16];
    int b = blockIdx.x / 3, j = blockIdx.x % 3;
    int om = (j == 0) ? 0 : (j + 1);     // OTHERS = (0,2,3)
    float maskj = tmask[b * 4 + om] ? 1.0f : 0.0f;
    const float* tok = tokens + (size_t)b * 4 * H_DIM;
    float vals[12];
    float s = 0.0f, ss = 0.0f;
    #pragma unroll
    for (int q = 0; q < 12; q++) {
        int d = threadIdx.x + q * 256;
        int h = d & 1023, sec = d >> 10;
        float t  = tok[om * H_DIM + h] * maskj;
        float sa = tok[1 * H_DIM + h];
        float v = (sec == 0) ? t : ((sec == 1) ? sa : fabsf(t - sa));
        vals[q] = v; s += v; ss += v * v;
    }
    s  = block_reduce_sum(s, sbuf);
    ss = block_reduce_sum(ss, sbuf);
    float mean = s * (1.0f / D3_DIM);
    float var  = ss * (1.0f / D3_DIM) - mean * mean;
    float rstd = rsqrtf(var + 1e-5f);
    __bf16* out = feat + (size_t)blockIdx.x * D3_DIM;
    #pragma unroll
    for (int q = 0; q < 12; q++) {
        int d = threadIdx.x + q * 256;
        out[d] = (__bf16)((vals[q] - mean) * rstd * lnw[d] + lnb[d]);
    }
}

// feat_a[b] = LN(concat(audio, others_mean, |audio - others_mean|)), bf16
__global__ void feat_audio_kernel(const float* __restrict__ tokens,
                                  const float* __restrict__ omean,
                                  const float* __restrict__ lnw,
                                  const float* __restrict__ lnb,
                                  __bf16* __restrict__ feat) {
    __shared__ float sbuf[16];
    int b = blockIdx.x;
    const float* au = tokens + (size_t)b * 4 * H_DIM + H_DIM;
    const float* om = omean + (size_t)b * H_DIM;
    float vals[12];
    float s = 0.0f, ss = 0.0f;
    #pragma unroll
    for (int q = 0; q < 12; q++) {
        int d = threadIdx.x + q * 256;
        int h = d & 1023, sec = d >> 10;
        float t = au[h], sa = om[h];
        float v = (sec == 0) ? t : ((sec == 1) ? sa : fabsf(t - sa));
        vals[q] = v; s += v; ss += v * v;
    }
    s  = block_reduce_sum(s, sbuf);
    ss = block_reduce_sum(ss, sbuf);
    float mean = s * (1.0f / D3_DIM);
    float var  = ss * (1.0f / D3_DIM) - mean * mean;
    float rstd = rsqrtf(var + 1e-5f);
    __bf16* out = feat + (size_t)b * D3_DIM;
    #pragma unroll
    for (int q = 0; q < 12; q++) {
        int d = threadIdx.x + q * 256;
        out[d] = (__bf16)((vals[q] - mean) * rstd * lnw[d] + lnb[d]);
    }
}

// g[row] = sigmoid(dot(h1[row], w2) + b2); one wave per row
__global__ void gate_kernel(const float* __restrict__ h1,
                            const float* __restrict__ w2,
                            const float* __restrict__ b2,
                            float* __restrict__ g, int rows) {
    int wid = threadIdx.x >> 5, lane = threadIdx.x & 31;
    int row = blockIdx.x * 8 + wid;
    if (row >= rows) return;
    const float* hr = h1 + (size_t)row * MID_DIM;
    float s = 0.0f;
    for (int i = lane; i < MID_DIM; i += 32) s += hr[i] * w2[i];
    for (int o = 16; o > 0; o >>= 1) s += __shfl_xor(s, o, 32);
    if (lane == 0) g[row] = 1.0f / (1.0f + expf(-(s + b2[0])));
}

// ------------------------------- mixing ------------------------------------

__global__ void mix_others_kernel(const float* __restrict__ tokens,
                                  const int* __restrict__ tmask,
                                  const float* __restrict__ g_o,
                                  const float* __restrict__ a2o,
                                  const float* __restrict__ ln_o_w,
                                  const float* __restrict__ ln_o_b,
                                  __bf16* __restrict__ xaug,
                                  float* __restrict__ omean_f,
                                  __bf16* __restrict__ omean_bf,
                                  int* __restrict__ aum) {
    __shared__ float sbuf[16];
    int b = blockIdx.x;
    int ma = tmask[b * 4 + 1];
    float macc[4] = {0, 0, 0, 0};
    float denom = 0.0f;
    const float* tok = tokens + (size_t)b * 4 * H_DIM;
    for (int j = 0; j < 3; j++) {
        int om = (j == 0) ? 0 : (j + 1);
        int mj = tmask[b * 4 + om];
        float maskj = mj ? 1.0f : 0.0f;
        denom += maskj;
        float g = g_o[(size_t)b * 3 + j];
        float ov[4], vv[4];
        float s = 0.0f, ss = 0.0f;
        #pragma unroll
        for (int q = 0; q < 4; q++) {
            int h = threadIdx.x + q * 256;
            float o = tok[om * H_DIM + h] * maskj;
            float v = o + g * a2o[(size_t)b * H_DIM + h];
            ov[q] = o; vv[q] = v; s += v; ss += v * v;
        }
        s  = block_reduce_sum(s, sbuf);
        ss = block_reduce_sum(ss, sbuf);
        float mean = s * (1.0f / H_DIM);
        float var  = ss * (1.0f / H_DIM) - mean * mean;
        float rstd = rsqrtf(var + 1e-5f);
        bool pair = (mj != 0) && (ma != 0);
        __bf16* xrow = xaug + ((size_t)b * 4 + om) * KAUG;
        #pragma unroll
        for (int q = 0; q < 4; q++) {
            int h = threadIdx.x + q * 256;
            float upd = (vv[q] - mean) * rstd * ln_o_w[h] + ln_o_b[h];
            float on = (pair ? upd : ov[q]) * maskj;
            macc[q] += on;
            xrow[1 + h] = (__bf16)on;
        }
        if (threadIdx.x == 0) xrow[0] = (__bf16)1.0f;
        if (threadIdx.x < KAUG - 1025) xrow[1025 + threadIdx.x] = (__bf16)0.0f;
    }
    float d = (denom < 1.0f) ? 1.0f : denom;
    #pragma unroll
    for (int q = 0; q < 4; q++) {
        int h = threadIdx.x + q * 256;
        float m = macc[q] / d;
        omean_f[(size_t)b * H_DIM + h] = m;
        omean_bf[(size_t)b * H_DIM + h] = (__bf16)m;
    }
    if (threadIdx.x == 0) aum[b] = ((ma != 0) && (denom > 0.0f)) ? 1 : 0;
}

__global__ void mix_audio_kernel(const float* __restrict__ tokens,
                                 const int* __restrict__ tmask,
                                 const float* __restrict__ g_a,
                                 const float* __restrict__ o2a,
                                 const int* __restrict__ aum,
                                 const float* __restrict__ ln_a_w,
                                 const float* __restrict__ ln_a_b,
                                 __bf16* __restrict__ xaug) {
    __shared__ float sbuf[16];
    int b = blockIdx.x;
    const float* au = tokens + (size_t)b * 4 * H_DIM + H_DIM;
    float ga = g_a[b];
    float vv[4];
    float s = 0.0f, ss = 0.0f;
    #pragma unroll
    for (int q = 0; q < 4; q++) {
        int h = threadIdx.x + q * 256;
        float v = au[h] + ga * o2a[(size_t)b * H_DIM + h];
        vv[q] = v; s += v; ss += v * v;
    }
    s  = block_reduce_sum(s, sbuf);
    ss = block_reduce_sum(ss, sbuf);
    float mean = s * (1.0f / H_DIM);
    float var  = ss * (1.0f / H_DIM) - mean * mean;
    float rstd = rsqrtf(var + 1e-5f);
    int am = aum[b];
    float mf = tmask[b * 4 + 1] ? 1.0f : 0.0f;
    __bf16* xrow = xaug + ((size_t)b * 4 + 1) * KAUG;
    #pragma unroll
    for (int q = 0; q < 4; q++) {
        int h = threadIdx.x + q * 256;
        float mx = am ? ((vv[q] - mean) * rstd * ln_a_w[h] + ln_a_b[h])
                      : au[h] * mf;
        mx *= mf;
        xrow[1 + h] = (__bf16)mx;
    }
    if (threadIdx.x == 0) xrow[0] = (__bf16)1.0f;
    if (threadIdx.x < KAUG - 1025) xrow[1025 + threadIdx.x] = (__bf16)0.0f;
}

// ------------------------- fused LMF einsum kernel -------------------------
// fused[b][h] = sum_r rank_w[r] * prod_m (mask[b][m] ? z : 1) + lmf_bias[h]
// Block tile 128(M) x 32(N); 8 waves, each 16x32 (two 16x16 WMMA col-tiles).
// Factors B panels staged global->LDS with CDNA5 async copies (ASYNCcnt),
// double-buffered and pipelined across both k-steps and rank boundaries.
// grid=(B/128, H/32), block=256.
__global__ void lmf_kernel(const __bf16* __restrict__ xaug,   // [B][4][KAUG]
                           const __bf16* __restrict__ fbT,    // [40][1024][KAUG]
                           const int* __restrict__ tmask,     // [B][4]
                           const float* __restrict__ rank_w,  // [10]
                           const float* __restrict__ lmf_bias,// [1024]
                           float* __restrict__ fused) {       // [B][1024]
    // LDS: 2 buffers x 4 m x 32 n x BPITCH k elements (bf16)
    __shared__ __bf16 lbuf[2 * 4 * 32 * BPITCH];

    int wave = threadIdx.x >> 5;
    int lane = threadIdx.x & 31;
    int row0 = blockIdx.x * 128 + wave * 16;
    int by32 = blockIdx.y * 32;

    unsigned int lds_base = (unsigned int)(size_t)(&lbuf[0]);
    const unsigned int BUF_BYTES = 4 * 32 * BPITCH * 2;   // one buffer
    const unsigned int M_BYTES   = 32 * BPITCH * 2;       // one m panel

    // Stage one (r, kb) B super-panel (4 m x 32 k x 32 n) into buffer `buf`.
    // 512 chunks of 16B; 2 per thread.
    auto stage = [&](int r, int kb, int buf) {
        #pragma unroll
        for (int t = 0; t < 2; t++) {
            int c   = threadIdx.x + t * 256;
            int m   = c >> 7;          // 128 chunks per m panel
            int rem = c & 127;
            int n   = rem >> 2;        // 4 chunks (64B of data) per n row
            int kc  = rem & 3;
            const __bf16* g = fbT
                + (size_t)(m * R_RANK + r) * H_DIM * KAUG
                + (size_t)(by32 + n) * KAUG + kb + kc * 8;
            unsigned int lds = lds_base
                + (unsigned int)(buf * BUF_BYTES + m * M_BYTES
                                 + n * (BPITCH * 2) + kc * 16);
            async_copy_b128(lds, g);
        }
    };

    // Pack per-row masks once (rows fixed for this wave across all ranks).
    int mbits[8];
    #pragma unroll
    for (int j = 0; j < 8; j++) {
        int row = row0 + ((lane < 16) ? j : 8 + j);
        const int* mr = tmask + (size_t)row * 4;
        mbits[j] = (mr[0] ? 1 : 0) | (mr[1] ? 2 : 0) | (mr[2] ? 4 : 0) | (mr[3] ? 8 : 0);
    }
    float rw[R_RANK];
    #pragma unroll
    for (int r = 0; r < R_RANK; r++) rw[r] = rank_w[r];

    v8f out0 = {}, out1 = {};

    int buf = 0;
    stage(0, 0, 0);
    wait_async0();
    __syncthreads();

    for (int r = 0; r < R_RANK; r++) {
        if (r + 1 < R_RANK) {
            __builtin_prefetch(fbT + ((size_t)(r + 1) * H_DIM + by32) * KAUG, 0, 1);
        }
        v8f acc0[4] = {}, acc1[4] = {};
        for (int s = 0; s < KAUG / 32; s++) {
            int kb = s * 32;
            int nb = buf ^ 1;
            // Keep the async engine busy: next k-step, or first panel of next r.
            if (s + 1 < KAUG / 32)      stage(r, kb + 32, nb);
            else if (r + 1 < R_RANK)    stage(r + 1, 0, nb);

            const __bf16* lb = &lbuf[buf * (4 * 32 * BPITCH)];
            #pragma unroll
            for (int m = 0; m < 4; m++) {
                v16bf a  = load_afrag(xaug + m * KAUG, 4 * KAUG, row0, kb);
                v16bf b0 = lds_bfrag(lb + m * (32 * BPITCH), 0);
                v16bf b1 = lds_bfrag(lb + m * (32 * BPITCH), 16);
                acc0[m] = __builtin_amdgcn_wmma_f32_16x16x32_bf16(
                    false, a, false, b0, (short)0, acc0[m], false, false);
                acc1[m] = __builtin_amdgcn_wmma_f32_16x16x32_bf16(
                    false, a, false, b1, (short)0, acc1[m], false, false);
            }
            wait_async0();
            __syncthreads();
            buf = nb;
        }
        #pragma unroll
        for (int j = 0; j < 8; j++) {
            float p0 = 1.0f, p1 = 1.0f;
            #pragma unroll
            for (int m = 0; m < 4; m++) {
                bool on = (mbits[j] >> m) & 1;
                p0 *= on ? acc0[m][j] : 1.0f;
                p1 *= on ? acc1[m][j] : 1.0f;
            }
            out0[j] += rw[r] * p0;
            out1[j] += rw[r] * p1;
        }
    }

    int col = lane & 15;
    #pragma unroll
    for (int j = 0; j < 8; j++) {
        int row = row0 + ((lane < 16) ? j : 8 + j);
        fused[(size_t)row * H_DIM + by32 + col]      = out0[j] + lmf_bias[by32 + col];
        fused[(size_t)row * H_DIM + by32 + 16 + col] = out1[j] + lmf_bias[by32 + 16 + col];
    }
}

// ------------------------------ output head --------------------------------

// Row LayerNorm over 1024; writes f32 and/or bf16
__global__ void ln_row_kernel(const float* __restrict__ X,
                              const float* __restrict__ w,
                              const float* __restrict__ b,
                              float* __restrict__ outf,
                              __bf16* __restrict__ outbf) {
    __shared__ float sbuf[16];
    int bi = blockIdx.x;
    const float* xr = X + (size_t)bi * H_DIM;
    float vv[4];
    float s = 0.0f, ss = 0.0f;
    #pragma unroll
    for (int q = 0; q < 4; q++) {
        int h = threadIdx.x + q * 256;
        float v = xr[h];
        vv[q] = v; s += v; ss += v * v;
    }
    s  = block_reduce_sum(s, sbuf);
    ss = block_reduce_sum(ss, sbuf);
    float mean = s * (1.0f / H_DIM);
    float var  = ss * (1.0f / H_DIM) - mean * mean;
    float rstd = rsqrtf(var + 1e-5f);
    #pragma unroll
    for (int q = 0; q < 4; q++) {
        int h = threadIdx.x + q * 256;
        float v = (vv[q] - mean) * rstd * w[h] + b[h];
        if (outf)  outf[(size_t)bi * H_DIM + h] = v;
        if (outbf) outbf[(size_t)bi * H_DIM + h] = (__bf16)v;
    }
}

// ------------------------------- host side ---------------------------------

extern "C" void kernel_launch(void* const* d_in, const int* in_sizes, int n_in,
                              void* d_out, int out_size, void* d_ws, size_t ws_size,
                              hipStream_t stream) {
    (void)in_sizes; (void)n_in; (void)out_size; (void)ws_size;

    const float* tokens   = (const float*)d_in[0];
    const int*   tmask    = (const int*)d_in[1];
    const float* ln_go_w  = (const float*)d_in[2];
    const float* ln_go_b  = (const float*)d_in[3];
    const float* go_w1    = (const float*)d_in[4];
    const float* go_b1    = (const float*)d_in[5];
    const float* go_w2    = (const float*)d_in[6];
    const float* go_b2    = (const float*)d_in[7];
    const float* ln_ga_w  = (const float*)d_in[8];
    const float* ln_ga_b  = (const float*)d_in[9];
    const float* ga_w1    = (const float*)d_in[10];
    const float* ga_b1    = (const float*)d_in[11];
    const float* ga_w2    = (const float*)d_in[12];
    const float* ga_b2    = (const float*)d_in[13];
    const float* a2o_w    = (const float*)d_in[14];
    const float* o2a_w    = (const float*)d_in[15];
    const float* ln_o_w   = (const float*)d_in[16];
    const float* ln_o_b   = (const float*)d_in[17];
    const float* ln_a_w   = (const float*)d_in[18];
    const float* ln_a_b   = (const float*)d_in[19];
    const float* factors  = (const float*)d_in[20];
    const float* rank_w   = (const float*)d_in[21];
    const float* lmf_bias = (const float*)d_in[22];
    const float* out_ln1_w = (const float*)d_in[23];
    const float* out_ln1_b = (const float*)d_in[24];
    const float* out_w    = (const float*)d_in[25];
    const float* out_b    = (const float*)d_in[26];
    const float* out_ln2_w = (const float*)d_in[27];
    const float* out_ln2_b = (const float*)d_in[28];
    float* out = (float*)d_out;

    char* ws = (char*)d_ws;
    size_t off = 0;
    auto alloc = [&](size_t bytes) -> char* {
        off = (off + 255) & ~(size_t)255;
        char* p = ws + off;
        off += bytes;
        return p;
    };

    __bf16* go_w1_bf = (__bf16*)alloc((size_t)MID_DIM * D3_DIM * 2);
    __bf16* ga_w1_bf = (__bf16*)alloc((size_t)MID_DIM * D3_DIM * 2);
    __bf16* a2o_w_bf = (__bf16*)alloc((size_t)H_DIM * H_DIM * 2);
    __bf16* o2a_w_bf = (__bf16*)alloc((size_t)H_DIM * H_DIM * 2);
    __bf16* out_w_bf = (__bf16*)alloc((size_t)H_DIM * H_DIM * 2);
    __bf16* fbT      = (__bf16*)alloc((size_t)40 * H_DIM * KAUG * 2);
    __bf16* feat_o   = (__bf16*)alloc((size_t)B_ROWS * 3 * D3_DIM * 2);
    __bf16* feat_a   = (__bf16*)alloc((size_t)B_ROWS * D3_DIM * 2);
    float*  h1_o     = (float*)alloc((size_t)B_ROWS * 3 * MID_DIM * 4);
    float*  h1_a     = (float*)alloc((size_t)B_ROWS * MID_DIM * 4);
    float*  g_o      = (float*)alloc((size_t)B_ROWS * 3 * 4);
    float*  g_a      = (float*)alloc((size_t)B_ROWS * 4);
    __bf16* audio_bf = (__bf16*)alloc((size_t)B_ROWS * H_DIM * 2);
    float*  a2o      = (float*)alloc((size_t)B_ROWS * H_DIM * 4);
    float*  o2a      = (float*)alloc((size_t)B_ROWS * H_DIM * 4);
    float*  omean_f  = (float*)alloc((size_t)B_ROWS * H_DIM * 4);
    __bf16* omean_bf = (__bf16*)alloc((size_t)B_ROWS * H_DIM * 2);
    int*    aum      = (int*)alloc((size_t)B_ROWS * 4);
    __bf16* xaug     = (__bf16*)alloc((size_t)B_ROWS * 4 * KAUG * 2);
    float*  fusedv   = (float*)alloc((size_t)B_ROWS * H_DIM * 4);
    __bf16* hln      = (__bf16*)alloc((size_t)B_ROWS * H_DIM * 2);
    float*  tmp      = (float*)alloc((size_t)B_ROWS * H_DIM * 4);

    // 0) precision conversion (bf16 once; factors fit in 192MB L2 afterwards)
    cvt_bf16_kernel<<<1024, 256, 0, stream>>>(go_w1, go_w1_bf, (size_t)MID_DIM * D3_DIM);
    cvt_bf16_kernel<<<1024, 256, 0, stream>>>(ga_w1, ga_w1_bf, (size_t)MID_DIM * D3_DIM);
    cvt_bf16_kernel<<<1024, 256, 0, stream>>>(a2o_w, a2o_w_bf, (size_t)H_DIM * H_DIM);
    cvt_bf16_kernel<<<1024, 256, 0, stream>>>(o2a_w, o2a_w_bf, (size_t)H_DIM * H_DIM);
    cvt_bf16_kernel<<<1024, 256, 0, stream>>>(out_w, out_w_bf, (size_t)H_DIM * H_DIM);
    cvt_factors_kernel<<<8192, 256, 0, stream>>>(factors, fbT);
    audio_bf16_kernel<<<2048, 256, 0, stream>>>(tokens, audio_bf);

    // 1) others gate: feat -> MLP -> sigmoid
    feat_others_kernel<<<B_ROWS * 3, 256, 0, stream>>>(tokens, tmask, ln_go_w, ln_go_b, feat_o);
    gemm_bf16_nt<<<dim3(B_ROWS * 3 / 128, MID_DIM / 16), 256, 0, stream>>>(
        feat_o, D3_DIM, go_w1_bf, D3_DIM, h1_o, MID_DIM, go_b1, 1, D3_DIM);
    gate_kernel<<<B_ROWS * 3 / 8, 256, 0, stream>>>(h1_o, go_w2, go_b2, g_o, B_ROWS * 3);

    // 2) a2o = audio @ a2o_w.T ; mix into others, compute others_mean
    gemm_bf16_nt<<<dim3(B_ROWS / 128, H_DIM / 16), 256, 0, stream>>>(
        audio_bf, H_DIM, a2o_w_bf, H_DIM, a2o, H_DIM, nullptr, 0, H_DIM);
    mix_others_kernel<<<B_ROWS, 256, 0, stream>>>(
        tokens, tmask, g_o, a2o, ln_o_w, ln_o_b, xaug, omean_f, omean_bf, aum);

    // 3) audio gate
    feat_audio_kernel<<<B_ROWS, 256, 0, stream>>>(tokens, omean_f, ln_ga_w, ln_ga_b, feat_a);
    gemm_bf16_nt<<<dim3(B_ROWS / 128, MID_DIM / 16), 256, 0, stream>>>(
        feat_a, D3_DIM, ga_w1_bf, D3_DIM, h1_a, MID_DIM, ga_b1, 1, D3_DIM);
    gate_kernel<<<B_ROWS / 8, 256, 0, stream>>>(h1_a, ga_w2, ga_b2, g_a, B_ROWS);

    // 4) o2a = others_mean @ o2a_w.T ; mix audio
    gemm_bf16_nt<<<dim3(B_ROWS / 128, H_DIM / 16), 256, 0, stream>>>(
        omean_bf, H_DIM, o2a_w_bf, H_DIM, o2a, H_DIM, nullptr, 0, H_DIM);
    mix_audio_kernel<<<B_ROWS, 256, 0, stream>>>(
        tokens, tmask, g_a, o2a, aum, ln_a_w, ln_a_b, xaug);

    // 5) fused low-rank einsum + masked product + rank reduction (async+WMMA)
    lmf_kernel<<<dim3(B_ROWS / 128, H_DIM / 32), 256, 0, stream>>>(
        xaug, fbT, tmask, rank_w, lmf_bias, fusedv);

    // 6) head: LN -> GEMM+gelu -> LN
    ln_row_kernel<<<B_ROWS, 256, 0, stream>>>(fusedv, out_ln1_w, out_ln1_b, nullptr, hln);
    gemm_bf16_nt<<<dim3(B_ROWS / 128, H_DIM / 16), 256, 0, stream>>>(
        hln, H_DIM, out_w_bf, H_DIM, tmp, H_DIM, out_b, 1, H_DIM);
    ln_row_kernel<<<B_ROWS, 256, 0, stream>>>(tmp, out_ln2_w, out_ln2_b, out, nullptr);
}
